// GPTQLinear_56788057587816
// MI455X (gfx1250) — compile-verified, matrix operand order checked by reading, not run on hardware
//
#include <hip/hip_runtime.h>

typedef __attribute__((ext_vector_type(16))) _Float16     v16h;
typedef __attribute__((ext_vector_type(8)))  float        v8f;
typedef __attribute__((ext_vector_type(4)))  _Float16     v4h;
typedef __attribute__((ext_vector_type(2)))  _Float16     v2h;
typedef __attribute__((ext_vector_type(4)))  unsigned int v4u;

constexpr int kInF   = 4096;
constexpr int kOutF  = 12288;
constexpr int BM = 256;         // block tile M
constexpr int BN = 256;         // block tile N
constexpr int BK = 64;          // block tile K (quant group 128 = 2 K-tiles)
constexpr int NT = 512;         // threads per block (16 waves, 4x4 wave grid)
constexpr int LDA = BK + 8;     // padded LDS stride (f16) for A [BM][LDA]
constexpr int LDB = BK + 8;     // padded LDS stride (f16) for B [BN][LDB], N-major

union FragH { v16h h; v4u u[2]; };

__global__ __launch_bounds__(NT)
void gptq_wmma_f16_kernel(const _Float16* __restrict__ x,
                          const int*      __restrict__ qweight,
                          const int*      __restrict__ qzeros,
                          const _Float16* __restrict__ scales,
                          _Float16*       __restrict__ y)
{
    __shared__ _Float16 smA[BM * LDA];   // 36.9 KB
    __shared__ _Float16 smB[BN * LDB];   // 36.9 KB (N-major: smB[n][k])

    const int tid   = threadIdx.x;
    const int lane  = tid & 31;
    const int wid   = tid >> 5;          // 0..15
    const int waveM = wid & 3;           // 0..3 -> 64-row slab
    const int waveN = wid >> 2;          // 0..3 -> 64-col slab

    const int nTile = blockIdx.x * BN;
    const int mTile = blockIdx.y * BM;

    v8f acc[4][4] = {};                  // 4(M) x 4(N) 16x16 f32 accumulators

    // ---- B-side thread mapping: 4 consecutive packed int32 per thread ----
    const int bi   = tid * 4;            // over (kp * 256 + n), n fastest
    const int b_n  = bi & (BN - 1);      // 0..252 step 4
    const int b_kp = bi >> 8;            // packed-K row in tile (0..7)
    const int gn0  = nTile + b_n;        // global column base (mult of 4)

    // ---- A-side thread mapping: 4 chunks of 8 f16 per thread ----
    const int a_row0 = tid >> 3;         // 0..63, + it*64
    const int a_col  = (tid & 7) * 8;

    // -------- register staging for software pipeline --------
    v4u aReg[4];
    v4u qReg;
    unsigned int zWord;
    v4h sReg;

    auto load_globals = [&](int kt) {
        #pragma unroll
        for (int it = 0; it < 4; ++it) {
            const int row = a_row0 + it * 64;
            aReg[it] = *(const v4u*)(x + (size_t)(mTile + row) * kInF + kt + a_col);
        }
        qReg = *(const v4u*)(qweight + (size_t)((kt >> 3) + b_kp) * kOutF + gn0);
        const int g = kt >> 7;           // quant group for this K-tile
        zWord = (unsigned int)qzeros[g * (kOutF / 8) + (gn0 >> 3)];
        sReg  = *(const v4h*)(scales + (size_t)g * kOutF + gn0);
    };

    load_globals(0);

    for (int kt = 0; kt < kInF; kt += BK) {
        // ---------------- LDS store phase (from staged regs) ----------------
        #pragma unroll
        for (int it = 0; it < 4; ++it)
            *(v4u*)(&smA[(a_row0 + it * 64) * LDA + a_col]) = aReg[it];

        const int zShift = (gn0 & 7) * 4;
        #pragma unroll
        for (int c = 0; c < 4; ++c) {
            const unsigned int q = qReg[c];
            const int z = (int)((zWord >> (zShift + 4 * c)) & 0xF);
            const _Float16 s  = sReg[c];
            const _Float16 c0 = -s * (_Float16)(short)(1024 + z);   // -(1024+z)*s
            const v2h s2 = {s, s};
            const v2h c2 = {c0, c0};

            // fp16 magic: (q | 0x6400) == 1024 + q  ->  w = h*s + c0 = (q - z)*s
            const unsigned int t0 = ( q        & 0x000F000Fu) | 0x64006400u; // k0,k4
            const unsigned int t1 = ((q >> 4)  & 0x000F000Fu) | 0x64006400u; // k1,k5
            const unsigned int t2 = ((q >> 8)  & 0x000F000Fu) | 0x64006400u; // k2,k6
            const unsigned int t3 = ((q >> 12) & 0x000F000Fu) | 0x64006400u; // k3,k7
            const v2h p0 = __builtin_bit_cast(v2h, t0) * s2 + c2;  // v_pk_fma_f16
            const v2h p1 = __builtin_bit_cast(v2h, t1) * s2 + c2;
            const v2h p2 = __builtin_bit_cast(v2h, t2) * s2 + c2;
            const v2h p3 = __builtin_bit_cast(v2h, t3) * s2 + c2;
            const unsigned int u0 = __builtin_bit_cast(unsigned int, p0);
            const unsigned int u1 = __builtin_bit_cast(unsigned int, p1);
            const unsigned int u2 = __builtin_bit_cast(unsigned int, p2);
            const unsigned int u3 = __builtin_bit_cast(unsigned int, p3);
            // repack (k0,k4)(k1,k5)(k2,k6)(k3,k7) -> k0..k7 contiguous
            v4u out;
            out[0] = (u0 & 0xFFFFu) | (u1 << 16);          // k0,k1
            out[1] = (u2 & 0xFFFFu) | (u3 << 16);          // k2,k3
            out[2] = (u0 >> 16)     | (u1 & 0xFFFF0000u);  // k4,k5
            out[3] = (u2 >> 16)     | (u3 & 0xFFFF0000u);  // k6,k7
            *(v4u*)(&smB[(b_n + c) * LDB + b_kp * 8]) = out;
        }

        __syncthreads();

        // issue next tile's global loads; latency hides under the WMMA stream
        if (kt + BK < kInF) load_globals(kt + BK);

        // ---------------- compute: 2 K-steps x (4A x 4B) = 32 WMMAs ----------------
        const int sel = (lane & 16) ? 8 : 0;   // lanes 16-31 hold K+8 (ISA layout)
        const int l15 = lane & 15;
        #pragma unroll
        for (int ks = 0; ks < BK; ks += 32) {
            FragH a[4], b[4];
            #pragma unroll
            for (int i = 0; i < 4; ++i) {
                const _Float16* pa = &smA[(waveM * 64 + i * 16 + l15) * LDA + ks + sel];
                a[i].u[0] = *(const v4u*)(pa);
                a[i].u[1] = *(const v4u*)(pa + 16);
            }
            #pragma unroll
            for (int j = 0; j < 4; ++j) {
                const _Float16* pb = &smB[(waveN * 64 + j * 16 + l15) * LDB + ks + sel];
                b[j].u[0] = *(const v4u*)(pb);
                b[j].u[1] = *(const v4u*)(pb + 16);
            }
            #pragma unroll
            for (int j = 0; j < 4; ++j) {
                #pragma unroll
                for (int i = 0; i < 4; ++i) {
                    acc[i][j] = __builtin_amdgcn_wmma_f32_16x16x32_f16(
                        false, a[i].h, false, b[j].h, (short)0, acc[i][j], false, false);
                }
            }
        }

        __syncthreads();
    }

    // ---------------- epilogue: C/D layout -> fp16 global stores ----------------
    const int l15 = lane & 15;
    const int hiM = (lane & 16) ? 8 : 0;   // lanes 16-31 hold rows M+8
    #pragma unroll
    for (int i = 0; i < 4; ++i) {
        #pragma unroll
        for (int j = 0; j < 4; ++j) {
            const int n = nTile + waveN * 64 + j * 16 + l15;
            #pragma unroll
            for (int r = 0; r < 8; ++r) {
                const int m = mTile + waveM * 64 + i * 16 + hiM + r;
                y[(size_t)m * kOutF + n] = (_Float16)acc[i][j][r];
            }
        }
    }
}

extern "C" void kernel_launch(void* const* d_in, const int* in_sizes, int n_in,
                              void* d_out, int out_size, void* d_ws, size_t ws_size,
                              hipStream_t stream) {
    const _Float16* x       = (const _Float16*)d_in[0];
    const int*      qweight = (const int*)d_in[1];
    const int*      qzeros  = (const int*)d_in[2];
    const _Float16* scales  = (const _Float16*)d_in[3];
    _Float16*       y       = (_Float16*)d_out;

    const int M = in_sizes[0] / kInF;            // tokens (8192)
    dim3 grid(kOutF / BN, M / BM);               // 48 x 32 workgroups
    gptq_wmma_f16_kernel<<<grid, NT, 0, stream>>>(x, qweight, qzeros, scales, y);
}
